// DANNet_51384988729941
// MI455X (gfx1250) — matrix operand order repaired and checked by reading, not compiled, not run on hardware
//
#include <hip/hip_runtime.h>
#include <hip/hip_bf16.h>

typedef __attribute__((ext_vector_type(16))) __bf16 v16bf;
typedef __attribute__((ext_vector_type(8)))  __bf16 v8bf;
typedef __attribute__((ext_vector_type(4)))  __bf16 v4bf;
typedef __attribute__((ext_vector_type(8)))  float  v8f;

#define BM 128
#define BN 64
#define BK 32
#define AP 40   // padded LDS row (bf16): 80B stride -> conflict-free 16-row strided b128 loads

// C[M,N] = act( (A[M,K] @ W[K,N]) * S + T ), V_WMMA_F32_16X16X32_BF16, f32 accum.
// MODE 0: +bias.  MODE 1: eval-mode BN folded to per-column scale/shift, then ReLU.
// perm: optional loop-invariant row gather for A (MoE routing). cnt: device effective M.
template<int MODE>
__global__ __launch_bounds__(256)
void gemm_wmma(const float* __restrict__ A, const float* __restrict__ W,
               float* __restrict__ C, int M, int N, int K,
               const float* __restrict__ bias,
               const float* __restrict__ bn_g, const float* __restrict__ bn_b,
               const float* __restrict__ bn_m, const float* __restrict__ bn_v,
               const int* __restrict__ perm, const int* __restrict__ cnt)
{
    __shared__ alignas(32) __bf16 As[2][BM][AP];
    __shared__ alignas(32) __bf16 Bs[2][BN][AP];

    const int m0 = blockIdx.y * BM;
    const int n0 = blockIdx.x * BN;
    int Meff = M;
    if (cnt) Meff = *cnt;
    if (m0 >= Meff) return;                 // uniform early-exit for routed tiles

    const int tid   = threadIdx.x;
    const int lane  = tid & 31;
    const int wave  = tid >> 5;
    const int wm    = (wave & 3) * 32;      // wave's 32-row strip
    const int wn    = (wave >> 2) * 32;     // wave's 32-col strip
    const int row16 = lane & 15;
    const int half  = lane >> 4;

    // ---- loop-invariant A staging descriptors: gather + bounds resolved ONCE (branch-free K-loop)
    const float* arow[4]; float amask[4]; int ar[4], akv[4];
    #pragma unroll
    for (int j = 0; j < 4; ++j) {
        int idx = tid + j * 256;            // 128 rows x 8 float4
        ar[j]  = idx >> 3;
        akv[j] = (idx & 7) << 2;
        int gr = m0 + ar[j];
        bool ok = (gr < Meff);
        int row = ok ? (perm ? perm[gr] : gr) : 0;
        arow[j]  = A + (size_t)row * K + akv[j];
        amask[j] = ok ? 1.f : 0.f;
    }
    // B staging: thread owns column bn_, K-groups bk0 and bk0+16 (K-contiguous -> ds_store_b64)
    const int bn_ = tid & 63;
    const int bk0 = (tid >> 6) << 2;
    const float* bcol = W + n0 + bn_;

    v8f acc00 = {}, acc01 = {}, acc10 = {}, acc11 = {};

    const int ntiles = K / BK;
    float4 pa[4];
    float  pb[8];

    // prologue: prefetch tile 0 into registers
    #pragma unroll
    for (int j = 0; j < 4; ++j) pa[j] = *(const float4*)(arow[j]);
    #pragma unroll
    for (int g = 0; g < 2; ++g)
        #pragma unroll
        for (int q = 0; q < 4; ++q)
            pb[g * 4 + q] = bcol[(size_t)(bk0 + g * 16 + q) * N];

    for (int t = 0; t < ntiles; ++t) {
        const int cur = t & 1;
        // commit prefetched tile to LDS buffer `cur` (bf16, packed stores)
        #pragma unroll
        for (int j = 0; j < 4; ++j) {
            v4bf v;
            v[0] = (__bf16)(pa[j].x * amask[j]);
            v[1] = (__bf16)(pa[j].y * amask[j]);
            v[2] = (__bf16)(pa[j].z * amask[j]);
            v[3] = (__bf16)(pa[j].w * amask[j]);
            *(v4bf*)&As[cur][ar[j]][akv[j]] = v;
        }
        #pragma unroll
        for (int g = 0; g < 2; ++g) {
            v4bf v;
            v[0] = (__bf16)pb[g * 4 + 0];
            v[1] = (__bf16)pb[g * 4 + 1];
            v[2] = (__bf16)pb[g * 4 + 2];
            v[3] = (__bf16)pb[g * 4 + 3];
            *(v4bf*)&Bs[cur][bn_][bk0 + g * 16] = v;
        }
        __syncthreads();

        // prefetch next tile (globals in flight behind the 4 WMMAs)
        if (t + 1 < ntiles) {
            const int kb = (t + 1) * BK;
            #pragma unroll
            for (int j = 0; j < 4; ++j) pa[j] = *(const float4*)(arow[j] + kb);
            #pragma unroll
            for (int g = 0; g < 2; ++g)
                #pragma unroll
                for (int q = 0; q < 4; ++q)
                    pb[g * 4 + q] = bcol[(size_t)(kb + bk0 + g * 16 + q) * N];
        }

        // fragments: A lane(row,half) holds K = 8h+[0..7] and 16+8h+[0..7]; B lane holds the same K
        v8bf a0lo = *(const v8bf*)&As[cur][wm + row16][8 * half];
        v8bf a0hi = *(const v8bf*)&As[cur][wm + row16][16 + 8 * half];
        v8bf a1lo = *(const v8bf*)&As[cur][wm + 16 + row16][8 * half];
        v8bf a1hi = *(const v8bf*)&As[cur][wm + 16 + row16][16 + 8 * half];
        v8bf b0lo = *(const v8bf*)&Bs[cur][wn + row16][8 * half];
        v8bf b0hi = *(const v8bf*)&Bs[cur][wn + row16][16 + 8 * half];
        v8bf b1lo = *(const v8bf*)&Bs[cur][wn + 16 + row16][8 * half];
        v8bf b1hi = *(const v8bf*)&Bs[cur][wn + 16 + row16][16 + 8 * half];
        v16bf a0, a1, b0, b1;
        #pragma unroll
        for (int i = 0; i < 8; ++i) {
            a0[i] = a0lo[i]; a0[i + 8] = a0hi[i];
            a1[i] = a1lo[i]; a1[i + 8] = a1hi[i];
            b0[i] = b0lo[i]; b0[i + 8] = b0hi[i];
            b1[i] = b1lo[i]; b1[i + 8] = b1hi[i];
        }

        acc00 = __builtin_amdgcn_wmma_f32_16x16x32_bf16(false, a0, false, b0, (short)0, acc00, false, false);
        acc01 = __builtin_amdgcn_wmma_f32_16x16x32_bf16(false, a0, false, b1, (short)0, acc01, false, false);
        acc10 = __builtin_amdgcn_wmma_f32_16x16x32_bf16(false, a1, false, b0, (short)0, acc10, false, false);
        acc11 = __builtin_amdgcn_wmma_f32_16x16x32_bf16(false, a1, false, b1, (short)0, acc11, false, false);
        // single barrier per K-step: next iteration writes the OTHER buffer, so only the
        // store(t)->compute(t) edge needs ordering.
    }

    // ---- epilogue: per-column affine (+ReLU); C/D layout: row = v + 8*half, col = lane%16
    const int col0 = n0 + wn + row16;
    const int col1 = col0 + 16;
    float S0 = 1.f, T0, S1 = 1.f, T1;
    if (MODE == 1) {
        float s0 = bn_g[col0] * rsqrtf(bn_v[col0] + 1e-5f);
        float s1 = bn_g[col1] * rsqrtf(bn_v[col1] + 1e-5f);
        S0 = s0; T0 = bias[col0] * s0 + (bn_b[col0] - bn_m[col0] * s0);
        S1 = s1; T1 = bias[col1] * s1 + (bn_b[col1] - bn_m[col1] * s1);
    } else {
        T0 = bias[col0]; T1 = bias[col1];
    }
    const int r0 = m0 + wm + 8 * half;
    #pragma unroll
    for (int v = 0; v < 8; ++v) {
        int ga = r0 + v, gb = r0 + 16 + v;
        if (ga < Meff) {
            float x0 = acc00[v] * S0 + T0;
            float x1 = acc01[v] * S1 + T1;
            if (MODE == 1) { x0 = fmaxf(x0, 0.f); x1 = fmaxf(x1, 0.f); }
            C[(size_t)ga * N + col0] = x0;
            C[(size_t)ga * N + col1] = x1;
        }
        if (gb < Meff) {
            float x0 = acc10[v] * S0 + T0;
            float x1 = acc11[v] * S1 + T1;
            if (MODE == 1) { x0 = fmaxf(x0, 0.f); x1 = fmaxf(x1, 0.f); }
            C[(size_t)gb * N + col0] = x0;
            C[(size_t)gb * N + col1] = x1;
        }
    }
}

// Row-wise LayerNorm + ReLU, in place; one 256-thread block per row; routed rows only.
__global__ __launch_bounds__(256)
void ln_relu(float* __restrict__ X, const float* __restrict__ g,
             const float* __restrict__ b, int N, const int* __restrict__ cnt)
{
    int row = blockIdx.x;
    if (cnt && row >= *cnt) return;
    float* x = X + (size_t)row * N;
    __shared__ float red[256];
    int tid = threadIdx.x;

    float s = 0.f;
    for (int i = tid; i < N; i += 256) s += x[i];
    red[tid] = s; __syncthreads();
    for (int off = 128; off > 0; off >>= 1) { if (tid < off) red[tid] += red[tid + off]; __syncthreads(); }
    float mu = red[0] / (float)N;
    __syncthreads();

    float v = 0.f;
    for (int i = tid; i < N; i += 256) { float d = x[i] - mu; v += d * d; }
    red[tid] = v; __syncthreads();
    for (int off = 128; off > 0; off >>= 1) { if (tid < off) red[tid] += red[tid + off]; __syncthreads(); }
    float rs = rsqrtf(red[0] / (float)N + 1e-5f);

    for (int i = tid; i < N; i += 256) {
        float y = (x[i] - mu) * rs * g[i] + b[i];
        x[i] = fmaxf(y, 0.f);
    }
}

// z2[B,128] @ fcw[128,8] + fcb -> log_softmax -> out; argmax builds routing (cnt, perm).
__global__ void head8(const float* __restrict__ Z, const float* __restrict__ Wf,
                      const float* __restrict__ bf, float* __restrict__ outS,
                      int* __restrict__ cnt, int* __restrict__ perm, int Btot, int K)
{
    int b = blockIdx.x * blockDim.x + threadIdx.x;
    if (b >= Btot) return;
    float acc[8];
    #pragma unroll
    for (int c = 0; c < 8; ++c) acc[c] = bf[c];
    const float* z = Z + (size_t)b * K;
    for (int k = 0; k < K; ++k) {
        float x = z[k];
        #pragma unroll
        for (int c = 0; c < 8; ++c) acc[c] += x * Wf[k * 8 + c];
    }
    float mx = acc[0]; int arg = 0;
    #pragma unroll
    for (int c = 1; c < 8; ++c) if (acc[c] > mx) { mx = acc[c]; arg = c; }  // first-max (jnp.argmax)
    float sum = 0.f;
    #pragma unroll
    for (int c = 0; c < 8; ++c) sum += __expf(acc[c] - mx);
    float lse = mx + __logf(sum);
    #pragma unroll
    for (int c = 0; c < 8; ++c) outS[b * 8 + c] = acc[c] - lse;
    int pos = atomicAdd(&cnt[arg], 1);
    perm[arg * Btot + pos] = b;
}

// H[B,K] @ W2[K,2] + b2 -> log_softmax -> outD (domain head).
__global__ void head2(const float* __restrict__ H, const float* __restrict__ W2,
                      const float* __restrict__ b2, float* __restrict__ outD, int Btot, int K)
{
    int b = blockIdx.x * blockDim.x + threadIdx.x;
    if (b >= Btot) return;
    const float* h = H + (size_t)b * K;
    float a0 = b2[0], a1 = b2[1];
    for (int k = 0; k < K; ++k) { float x = h[k]; a0 += x * W2[k * 2]; a1 += x * W2[k * 2 + 1]; }
    float m = fmaxf(a0, a1);
    float lse = m + __logf(__expf(a0 - m) + __expf(a1 - m));
    outD[b * 2] = a0 - lse;
    outD[b * 2 + 1] = a1 - lse;
}

// Expert head: compacted rows i < *cnt, scatter to original batch rows via perm.
__global__ void head2_scatter(const float* __restrict__ H, const float* __restrict__ W2,
                              const float* __restrict__ b2, const int* __restrict__ perm,
                              const int* __restrict__ cnt, float* __restrict__ outS, int K)
{
    int i = blockIdx.x * blockDim.x + threadIdx.x;
    if (i >= *cnt) return;
    const float* h = H + (size_t)i * K;
    float a0 = b2[0], a1 = b2[1];
    for (int k = 0; k < K; ++k) { float x = h[k]; a0 += x * W2[k * 2]; a1 += x * W2[k * 2 + 1]; }
    float m = fmaxf(a0, a1);
    float lse = m + __logf(__expf(a0 - m) + __expf(a1 - m));
    int b = perm[i];
    outS[b * 2] = a0 - lse;
    outS[b * 2 + 1] = a1 - lse;
}

__global__ void zero_cnt(int* cnt) { if (threadIdx.x < 8) cnt[threadIdx.x] = 0; }

extern "C" void kernel_launch(void* const* d_in, const int* in_sizes, int n_in,
                              void* d_out, int out_size, void* d_ws, size_t ws_size,
                              hipStream_t stream)
{
    const int B_ = 4096, D_ = 512, E_ = 8;
    auto F = [&](int i) { return (const float*)d_in[i]; };

    const float* source = F(0);
    float* out     = (float*)d_out;
    float* out_src = out;                 // [4096, 8]
    float* out_d   = out + 4096 * 8;      // [4096, 2]
    float* out_s   = out + 4096 * 10;     // [4096, 2]
    float* out_sh  = out + 4096 * 12;     // [4096, 512]

    char* ws = (char*)d_ws;
    float* z1 = (float*)(ws);                              //  4 MiB  [B,256]
    float* z2 = (float*)(ws + ( 4ull << 20));              //  2 MiB  [B,128]
    float* h1 = (float*)(ws + ( 6ull << 20));              // 16 MiB  [B,1024]
    float* h2 = (float*)(ws + (22ull << 20));              // 16 MiB  [B,1024]
    float* h3 = (float*)(ws + (38ull << 20));              //  8 MiB  [B,512]
    int*   cnt  = (int*)(ws + (46ull << 20));              //  8 ints
    int*   perm = (int*)(ws + (46ull << 20) + 256);        //  E*B ints
    float* e1 = h1; float* e2 = h2; float* e3 = h3;        // experts reuse (domain done first)

    dim3 blk(256);
    auto grd = [](int M, int N) { return dim3((unsigned)(N / BN), (unsigned)(M / BM)); };

    // output 4: grad-reverse is forward-identity -> copy features
    hipMemcpyAsync(out_sh, source, (size_t)B_ * D_ * sizeof(float),
                   hipMemcpyDeviceToDevice, stream);
    zero_cnt<<<1, 32, 0, stream>>>(cnt);

    // bottleneck + source classifier (builds routing)
    gemm_wmma<1><<<grd(B_, 256), blk, 0, stream>>>(source, F(1), z1, B_, 256, 512,
        F(2), F(3), F(4), F(5), F(6), nullptr, nullptr);
    gemm_wmma<1><<<grd(B_, 128), blk, 0, stream>>>(z1, F(7), z2, B_, 128, 256,
        F(8), F(9), F(10), F(11), F(12), nullptr, nullptr);
    head8<<<B_ / 256, 256, 0, stream>>>(z2, F(13), F(14), out_src, cnt, perm, B_, 128);

    // domain classifier
    gemm_wmma<1><<<grd(B_, 1024), blk, 0, stream>>>(source, F(15), h1, B_, 1024, 512,
        F(16), F(17), F(18), F(19), F(20), nullptr, nullptr);
    gemm_wmma<1><<<grd(B_, 1024), blk, 0, stream>>>(h1, F(21), h2, B_, 1024, 1024,
        F(22), F(23), F(24), F(25), F(26), nullptr, nullptr);
    gemm_wmma<1><<<grd(B_, 512), blk, 0, stream>>>(h2, F(27), h3, B_, 512, 1024,
        F(28), F(29), F(30), F(31), F(32), nullptr, nullptr);
    head2<<<B_ / 256, 256, 0, stream>>>(h3, F(33), F(34), out_d, B_, 512);

    // routed experts: each processes only its count[e] rows (fixed grids, device-side early exit)
    for (int e = 0; e < E_; ++e) {
        const int*   pe   = perm + (size_t)e * B_;
        const int*   ce   = cnt + e;
        const float* ew1e = F(35) + (size_t)e * 512 * 1024;
        const float* ew2e = F(39) + (size_t)e * 1024 * 1024;
        const float* ew3e = F(43) + (size_t)e * 1024 * 512;
        const float* ew4e = F(47) + (size_t)e * 512 * 2;

        gemm_wmma<0><<<grd(B_, 1024), blk, 0, stream>>>(source, ew1e, e1, B_, 1024, 512,
            F(36) + e * 1024, nullptr, nullptr, nullptr, nullptr, pe, ce);
        ln_relu<<<B_, 256, 0, stream>>>(e1, F(37) + e * 1024, F(38) + e * 1024, 1024, ce);

        gemm_wmma<0><<<grd(B_, 1024), blk, 0, stream>>>(e1, ew2e, e2, B_, 1024, 1024,
            F(40) + e * 1024, nullptr, nullptr, nullptr, nullptr, nullptr, ce);
        ln_relu<<<B_, 256, 0, stream>>>(e2, F(41) + e * 1024, F(42) + e * 1024, 1024, ce);

        gemm_wmma<0><<<grd(B_, 512), blk, 0, stream>>>(e2, ew3e, e3, B_, 512, 1024,
            F(44) + e * 512, nullptr, nullptr, nullptr, nullptr, nullptr, ce);
        ln_relu<<<B_, 256, 0, stream>>>(e3, F(45) + e * 512, F(46) + e * 512, 512, ce);

        head2_scatter<<<B_ / 256, 256, 0, stream>>>(e3, ew4e, F(48) + e * 2, pe, ce, out_s, 512);
    }
}
